// DeformableAdaptiveShift_15891378995295
// MI455X (gfx1250) — matrix operand order repaired and checked by reading, not compiled, not run on hardware
//
#include <hip/hip_runtime.h>
#include <hip/hip_bf16.h>

typedef __attribute__((ext_vector_type(16))) _Float16 v16h;
typedef __attribute__((ext_vector_type(8)))  float    v8f;

#define HH   64
#define WWID 64
#define HWW  4096
#define BATCH 4

__device__ __forceinline__ float gelu_f(float v) {
  return 0.5f * v * (1.0f + erff(v * 0.70710678118654752440f));
}
__device__ __forceinline__ float sigmoid_f(float v) {
  return 1.0f / (1.0f + expf(-v));
}

// ---------------------------------------------------------------------------
// Transpose + fp16 conversion: x [B, HW, C=256] f32 -> xh [B, C, HW] f16
// ---------------------------------------------------------------------------
__global__ void pack_x_kernel(const float* __restrict__ x, _Float16* __restrict__ xh) {
  __shared__ float tile[32][33];
  int bx = blockIdx.x;  // spatial tile (HW/32)
  int by = blockIdx.y;  // channel tile (256/32)
  int b  = blockIdx.z;
  int tn = threadIdx.x; // 0..31
  int tm = threadIdx.y; // 0..7
#pragma unroll
  for (int i = 0; i < 4; ++i) {
    int n = bx * 32 + tm + i * 8;
    tile[tm + i * 8][tn] = x[((size_t)b * HWW + n) * 256 + by * 32 + tn];
  }
  __syncthreads();
#pragma unroll
  for (int i = 0; i < 4; ++i) {
    int cc = tm + i * 8;
    xh[((size_t)b * 256 + by * 32 + cc) * HWW + bx * 32 + tn] = (_Float16)tile[tn][cc];
  }
}

// ---------------------------------------------------------------------------
// Repack conv weights [Cout, Cin_g, 3, 3] f32 into WMMA A-fragment order.
//   oc = octile*16 + (lane & 15)
//   K  = j + (lane>=16 ? 16 : 0)     (contiguous k order, matches B frag)
//   ic = kb*32 + K
// Zero-padded when oc >= Cout or ic >= Cin_g.
// ---------------------------------------------------------------------------
__global__ void pack_w_kernel(const float* __restrict__ w, _Float16* __restrict__ wp,
                              int Cout, int Cin_g, int noctile, int nkb) {
  size_t total = (size_t)noctile * 9 * nkb * 32 * 16;
  size_t i = (size_t)blockIdx.x * 256 + threadIdx.x;
  if (i >= total) return;
  int j    = (int)(i & 15);
  int lane = (int)((i >> 4) & 31);
  size_t rest = i >> 9;
  int kb  = (int)(rest % nkb); rest /= nkb;
  int tap = (int)(rest % 9);
  int octile = (int)(rest / 9);
  int K  = j + ((lane >= 16) ? 16 : 0);
  int ic = kb * 32 + K;
  int oc = octile * 16 + (lane & 15);
  float val = 0.0f;
  if (oc < Cout && ic < Cin_g)
    val = w[((size_t)oc * Cin_g + ic) * 9 + tap];
  wp[i] = (_Float16)val;
}

// ---------------------------------------------------------------------------
// Implicit-GEMM 3x3 SAME conv via v_wmma_f32_16x16x32_f16.
// Workgroup: 4 waves = 64 out-channels x 64 spatial (one image row).
// kh-outer loop: each staged 32ic x 64px tile feeds all 3 kw taps
// (12 WMMAs per stage). Double-buffered LDS + register preload pipeline,
// global_prefetch two stages ahead.
// ACT: 0 = none (f32 out), 1 = GELU (f16 out), 2 = sigmoid (f32 out).
// ---------------------------------------------------------------------------
template <int ACT>
__global__ __launch_bounds__(128)
void conv_wmma_kernel(const _Float16* __restrict__ in, const _Float16* __restrict__ wp,
                      const float* __restrict__ bias, void* __restrict__ outv,
                      int Cin_total, int Cin_g, int Cout, int grouped, int nkb) {
  __shared__ _Float16 lbuf[2][64][32];  // double-buffered [spatial n][k]

  const int h    = blockIdx.x;        // image row
  const int gy   = blockIdx.y;        // 64-oc tile (== group for grouped conv)
  const int b    = blockIdx.z;
  const int tid  = threadIdx.x;
  const int wave = tid >> 5;
  const int lane = tid & 31;
  const int octile = gy * 4 + wave;
  const int in_ch_base = grouped ? gy * Cin_g : 0;

  v8f acc[4];
#pragma unroll
  for (int q = 0; q < 4; ++q)
#pragma unroll
    for (int r = 0; r < 8; ++r) acc[q][r] = 0.0f;

  const int koff = (lane & 16);       // which K-half this lane's B frag covers
  const int ncol = lane & 15;
  const int sic  = tid >> 2;          // staging: input channel 0..31
  const int sn0  = (tid & 3) << 4;    // staging: spatial base 0/16/32/48

  // Valid kh range is contiguous: rows h-1..h+1 clipped to [0, H).
  const int kh_lo = (h == 0) ? 1 : 0;
  const int kh_hi = (h == HH - 1) ? 1 : 2;
  const int nvalid  = kh_hi - kh_lo + 1;
  const int nstages = nvalid * nkb;

  // Per-thread global base for staging reads.
  const _Float16* base =
      in + ((size_t)b * Cin_total + in_ch_base + sic) * HWW + sn0;

  // Stage s = hv*nkb + kb  ->  row (h + kh_lo + hv - 1), channel block kb.
  v16h t = *(const v16h*)(base + (size_t)(h + kh_lo - 1) * WWID);  // s = 0
  int hv = 0, kb = 0, buf = 0;

  for (int s = 0; s < nstages; ++s) {
    const int kh = kh_lo + hv;

    __syncthreads();  // lbuf[buf] readers from 2 stages ago are done
#pragma unroll
    for (int i = 0; i < 16; ++i) lbuf[buf][sn0 + i][sic] = t[i];
    __syncthreads();

    // Pipeline: preload stage s+1 into registers, prefetch stage s+2 to cache.
    int kb1 = kb + 1, hv1 = hv;
    if (kb1 == nkb) { kb1 = 0; hv1 = hv + 1; }
    if (s + 1 < nstages)
      t = *(const v16h*)(base + (size_t)kb1 * 32 * HWW +
                         (size_t)(h + kh_lo + hv1 - 1) * WWID);
    int kb2 = kb1 + 1, hv2 = hv1;
    if (kb2 == nkb) { kb2 = 0; hv2 = hv1 + 1; }
    if (s + 2 < nstages)
      __builtin_prefetch((const void*)(base + (size_t)kb2 * 32 * HWW +
                                       (size_t)(h + kh_lo + hv2 - 1) * WWID),
                         0, 1);

    // 3 kw taps x 4 accumulators = 12 WMMAs on this tile.
#pragma unroll
    for (int kw = 0; kw < 3; ++kw) {
      const int tap = kh * 3 + kw;
      v16h a = ((const v16h*)wp)[(size_t)((octile * 9 + tap) * nkb + kb) * 32 + lane];
#pragma unroll
      for (int q = 0; q < 4; ++q) {
        const int np = q * 16 + ncol + kw - 1;   // shifted input column
        v16h bf;
#pragma unroll
        for (int i = 0; i < 16; ++i) bf[i] = (_Float16)0.0f;
        if ((unsigned)np < 64u) bf = *(const v16h*)(&lbuf[buf][np][koff]);
        acc[q] = __builtin_amdgcn_wmma_f32_16x16x32_f16(
            false, a, false, bf, (short)0, acc[q], false, false);
      }
    }
    hv = hv1; kb = kb1; buf ^= 1;
  }

  // Epilogue: D layout — lanes 0-15: N=lane, M=r; lanes 16-31: N=lane-16, M=8+r.
  const int mbase = (lane & 16) ? 8 : 0;
  _Float16* out16 = (_Float16*)outv;
  float*    out32 = (float*)outv;
#pragma unroll
  for (int q = 0; q < 4; ++q) {
    const int s = h * WWID + q * 16 + ncol;
#pragma unroll
    for (int r = 0; r < 8; ++r) {
      const int oc = octile * 16 + mbase + r;
      if (oc < Cout) {
        float val = acc[q][r] + bias[oc];
        size_t idx = ((size_t)b * Cout + oc) * HWW + s;
        if (ACT == 1)      out16[idx] = (_Float16)gelu_f(val);
        else if (ACT == 2) out32[idx] = sigmoid_f(val);
        else               out32[idx] = val;
      }
    }
  }
}

// ---------------------------------------------------------------------------
// Deformable bilinear sampling + modulation + mean over P=9 points.
// One block per (b, pixel); 256 threads = channels. The 72+36 per-pixel
// offset/modulation scalars are staged into LDS once per block; the 4
// bilinear gathers stay channel-contiguous (L2-resident).
// ---------------------------------------------------------------------------
__global__ __launch_bounds__(256)
void sample_kernel(const float* __restrict__ x, const float* __restrict__ off,
                   const float* __restrict__ mod, float* __restrict__ out) {
  __shared__ float soff[72];   // [g*18 + d*9 + p]
  __shared__ float smod[36];   // [g*9 + p]
  const int n = blockIdx.x;    // pixel
  const int b = blockIdx.y;
  const int c = threadIdx.x;
  const int g = c >> 6;
  const int h = n >> 6;
  const int w = n & 63;

  if (c < 72)                soff[c]      = off[((size_t)b * 72 + c) * HWW + n];
  else if (c < 108)          smod[c - 72] = mod[((size_t)b * 36 + (c - 72)) * HWW + n];
  __syncthreads();

  const float* xb = x + (size_t)b * HWW * 256;
  float acc = 0.0f;
#pragma unroll
  for (int p = 0; p < 9; ++p) {
    const float ox = soff[g * 18 + p];
    const float oy = soff[g * 18 + 9 + p];
    const float mv = smod[g * 9 + p];
    float px = fminf(fmaxf((float)w + (float)(p % 3 - 1) + ox, 0.0f), 63.0f);
    float py = fminf(fmaxf((float)h + (float)(p / 3 - 1) + oy, 0.0f), 63.0f);
    float x0f = floorf(px), y0f = floorf(py);
    int x0 = (int)x0f, y0 = (int)y0f;
    int x1 = min(x0 + 1, 63), y1 = min(y0 + 1, 63);
    float wx = px - x0f, wy = py - y0f;
    float v00 = xb[(size_t)(y0 * 64 + x0) * 256 + c];
    float v01 = xb[(size_t)(y0 * 64 + x1) * 256 + c];
    float v10 = xb[(size_t)(y1 * 64 + x0) * 256 + c];
    float v11 = xb[(size_t)(y1 * 64 + x1) * 256 + c];
    float v = v00 * (1.0f - wx) * (1.0f - wy) + v01 * wx * (1.0f - wy) +
              v10 * (1.0f - wx) * wy + v11 * wx * wy;
    acc += v * mv;
  }
  out[((size_t)b * HWW + n) * 256 + c] = acc * (1.0f / 9.0f);
}

// ---------------------------------------------------------------------------
extern "C" void kernel_launch(void* const* d_in, const int* in_sizes, int n_in,
                              void* d_out, int out_size, void* d_ws, size_t ws_size,
                              hipStream_t stream) {
  (void)in_sizes; (void)n_in; (void)out_size; (void)ws_size;
  const float* x      = (const float*)d_in[0];
  const float* off_w1 = (const float*)d_in[1];
  const float* off_b1 = (const float*)d_in[2];
  const float* off_w2 = (const float*)d_in[3];
  const float* off_b2 = (const float*)d_in[4];
  const float* mod_w1 = (const float*)d_in[5];
  const float* mod_b1 = (const float*)d_in[6];
  const float* mod_w2 = (const float*)d_in[7];
  const float* mod_b2 = (const float*)d_in[8];
  float* out = (float*)d_out;

  char* ws = (char*)d_ws;
  size_t pos = 0;
  auto take = [&](size_t bytes) -> void* {
    void* p = (void*)(ws + pos);
    pos += (bytes + 255) & ~(size_t)255;
    return p;
  };
  _Float16* xh   = (_Float16*)take((size_t)BATCH * 256 * HWW * 2);  // x f16 NCHW
  _Float16* h1   = (_Float16*)take((size_t)BATCH * 256 * HWW * 2);  // gelu(conv1)
  _Float16* m1   = (_Float16*)take((size_t)BATCH * 64  * HWW * 2);  // gelu(mconv1)
  float*    offb = (float*)take((size_t)BATCH * 72 * HWW * 4);      // raw offsets
  float*    modb = (float*)take((size_t)BATCH * 36 * HWW * 4);      // sigmoid mod
  _Float16* wp1  = (_Float16*)take((size_t)16 * 9 * 2 * 512 * 2);   // 147456 frag elems
  _Float16* wp2  = (_Float16*)take((size_t)8  * 9 * 8 * 512 * 2);   // 294912
  _Float16* wp3  = (_Float16*)take((size_t)4  * 9 * 8 * 512 * 2);   // 147456
  _Float16* wp4  = (_Float16*)take((size_t)4  * 9 * 2 * 512 * 2);   // 36864

  // Stage 0: layout / precision transforms.
  pack_x_kernel<<<dim3(HWW / 32, 8, BATCH), dim3(32, 8, 1), 0, stream>>>(x, xh);
  pack_w_kernel<<<(147456 + 255) / 256, 256, 0, stream>>>(off_w1, wp1, 256, 64, 16, 2);
  pack_w_kernel<<<(294912 + 255) / 256, 256, 0, stream>>>(off_w2, wp2, 72, 256, 8, 8);
  pack_w_kernel<<<(147456 + 255) / 256, 256, 0, stream>>>(mod_w1, wp3, 64, 256, 4, 8);
  pack_w_kernel<<<(36864  + 255) / 256, 256, 0, stream>>>(mod_w2, wp4, 36, 64, 4, 2);

  // Stage 1: WMMA convs.
  // conv1 (grouped, G=4): Cg_in=64 -> nkb=2; gy == group == 64-oc tile.
  conv_wmma_kernel<1><<<dim3(HH, 4, BATCH), 128, 0, stream>>>(
      xh, wp1, off_b1, (void*)h1, 256, 64, 256, 1, 2);
  // conv2 (dense, 72 oc): Cin=256 -> nkb=8.
  conv_wmma_kernel<0><<<dim3(HH, 2, BATCH), 128, 0, stream>>>(
      h1, wp2, off_b2, (void*)offb, 256, 256, 72, 0, 8);
  // mod conv1 (dense, 64 oc): Cin=256 -> nkb=8.
  conv_wmma_kernel<1><<<dim3(HH, 1, BATCH), 128, 0, stream>>>(
      xh, wp3, mod_b1, (void*)m1, 256, 256, 64, 0, 8);
  // mod conv2 (dense, 36 oc): Cin=64 -> nkb=2, sigmoid epilogue.
  conv_wmma_kernel<2><<<dim3(HH, 1, BATCH), 128, 0, stream>>>(
      m1, wp4, mod_b2, (void*)modb, 64, 64, 36, 0, 2);

  // Stage 2: deformable sampling (L2-resident gathers).
  sample_kernel<<<dim3(HWW, BATCH, 1), 256, 0, stream>>>(x, offb, modb, out);
}
